// MT5Attention_80951543595303
// MI455X (gfx1250) — matrix operand order repaired and checked by reading, not compiled, not run on hardware
//
#include <hip/hip_runtime.h>
#include <math.h>

// ---------------------------------------------------------------------------
// MT5 attention for gfx1250 (CDNA5): bf16 WMMA pipeline + TDM tensor load.
// ---------------------------------------------------------------------------
constexpr int kB  = 4;
constexpr int kS  = 2048;
constexpr int kH  = 16;
constexpr int kHD = 64;
constexpr int kDM = 1024;

typedef __attribute__((ext_vector_type(16))) __bf16   v16bf;
typedef __attribute__((ext_vector_type(8)))  float    v8f;
typedef __attribute__((ext_vector_type(4)))  unsigned v4u;
typedef __attribute__((ext_vector_type(8)))  int      v8i;
typedef __attribute__((ext_vector_type(4)))  int      v4i;

static __device__ __forceinline__ __bf16 f2bf(float f) { return (__bf16)f; }

static __device__ __forceinline__ v8f wmma_bf16(v16bf a, v16bf b, v8f c) {
  // (neg_a, A, neg_b, B, c_mod, C, reuse_a, reuse_b)
  return __builtin_amdgcn_wmma_f32_16x16x32_bf16(false, a, false, b, (short)0, c,
                                                 false, false);
}

// A-matrix fragment (16x32 bf16, MxK), memory row-major [m][k], stride lda.
// ISA layout: lanes 0-15 -> M=lane (lanes 16-31 same M, K offset +8);
// VGPR v in {0..7}: K = (v>=4)*16 + half*8 + (v&3)*2 + {0,1}.
static __device__ __forceinline__ v16bf load_a_bf16(const __bf16* __restrict__ A,
                                                    int lda) {
  const int l = threadIdx.x & 31;
  const int m = l & 15;
  const int half = l >> 4;
  v16bf a;
#pragma unroll
  for (int e = 0; e < 16; e += 2) {
    const int k = ((e >> 3) << 4) + (half << 3) + (((e >> 1) & 3) << 1);
    a[e]     = A[m * lda + k];
    a[e + 1] = A[m * lda + k + 1];
  }
  return a;
}

// Same A fragment but sourced from f32 memory (LDS or global), converted.
static __device__ __forceinline__ v16bf load_a_f32(const float* A, int lda) {
  const int l = threadIdx.x & 31;
  const int m = l & 15;
  const int half = l >> 4;
  v16bf a;
#pragma unroll
  for (int e = 0; e < 16; ++e) {
    const int k = ((e >> 3) << 4) + (half << 3) + (((e >> 1) & 3) << 1) + (e & 1);
    a[e] = f2bf(A[m * lda + k]);
  }
  return a;
}

// B-matrix fragment (32x16 bf16, KxN) from memory holding B^T row-major
// (k contiguous, stride ldb). Lane = N; element e: K = (e>=8)*16 + half*8 + (e&7).
static __device__ __forceinline__ v16bf load_bt_bf16(const __bf16* __restrict__ Bt,
                                                     int ldb) {
  const int l = threadIdx.x & 31;
  const int n = l & 15;
  const int half = l >> 4;
  v16bf b;
#pragma unroll
  for (int e = 0; e < 16; ++e) b[e] = Bt[n * ldb + ((e >> 3) << 4) + (half << 3) + (e & 7)];
  return b;
}

// Relative position bucket (bidirectional, nb=32, max_distance=128).
static __device__ __forceinline__ int rel_bucket(int rel) {
  int bucket = (rel > 0) ? 16 : 0;
  const int rp = rel < 0 ? -rel : rel;
  if (rp < 8) return bucket + rp;
  int v = 8 + (int)(__logf((float)rp * 0.125f) * (8.0f / __logf(16.0f)));
  return bucket + (v < 15 ? v : 15);
}

// ---------------------------------------------------------------------------
// 1) f32 -> bf16 elementwise convert (hidden_states)
// ---------------------------------------------------------------------------
__global__ void k_cvt_bf16(const float* __restrict__ in, __bf16* __restrict__ out,
                           int n) {
  int i = blockIdx.x * blockDim.x + threadIdx.x;
  if (i < n) out[i] = f2bf(in[i]);
}

// ---------------------------------------------------------------------------
// 2) Transpose + convert a 1024x1024 weight: Wt[o][i] = (bf16)W[i][o]
// ---------------------------------------------------------------------------
__global__ void k_wt_bf16(const float* __restrict__ W, __bf16* __restrict__ Wt) {
  __shared__ __bf16 tile[32][33];
  const int tx = threadIdx.x;
  const int ty = threadIdx.y;
  const int i0 = blockIdx.y * 32;
  const int o0 = blockIdx.x * 32;
#pragma unroll
  for (int s = 0; s < 32; s += 8)
    tile[ty + s][tx] = f2bf(W[(size_t)(i0 + ty + s) * kDM + (o0 + tx)]);
  __syncthreads();
#pragma unroll
  for (int s = 0; s < 32; s += 8)
    Wt[(size_t)(o0 + ty + s) * kDM + (i0 + tx)] = tile[tx][ty + s];
}

// ---------------------------------------------------------------------------
// 3) QKV projection: X[8192x1024] @ W -> Q,K as [b,h,s,d] bf16, V as [b,h,d,s].
// ---------------------------------------------------------------------------
__global__ void k_qkv(const __bf16* __restrict__ Xb, const __bf16* __restrict__ Wqt,
                      const __bf16* __restrict__ Wkt, const __bf16* __restrict__ Wvt,
                      __bf16* __restrict__ Qb, __bf16* __restrict__ Kb,
                      __bf16* __restrict__ Vt) {
  const int n0 = blockIdx.x * 16;
  const int m0 = blockIdx.y * 16;
  const int which = blockIdx.z;            // 0=Q 1=K 2=V
  const __bf16* Wt = which == 0 ? Wqt : (which == 1 ? Wkt : Wvt);
  v8f acc = {};
  for (int k0 = 0; k0 < kDM; k0 += 32) {
    v16bf a = load_a_bf16(Xb + (size_t)m0 * kDM + k0, kDM);
    v16bf b = load_bt_bf16(Wt + (size_t)n0 * kDM + k0, kDM);
    acc = wmma_bf16(a, b, acc);
  }
  const int l = threadIdx.x & 31;
  const int nn = n0 + (l & 15);
  const int h = nn >> 6, d = nn & (kHD - 1);
#pragma unroll
  for (int r = 0; r < 8; ++r) {
    const int mm = m0 + r + ((l >> 4) << 3);
    const int bb = mm >> 11, s = mm & (kS - 1);
    const size_t bh = (size_t)bb * kH + h;
    const __bf16 val = f2bf(acc[r]);
    if (which == 0)      Qb[(bh * kS + s) * kHD + d] = val;
    else if (which == 1) Kb[(bh * kS + s) * kHD + d] = val;
    else                 Vt[(bh * kHD + d) * kS + s] = val;   // transposed
  }
}

// ---------------------------------------------------------------------------
// 4) Relative position bias -> d_out slice [H, S, S]
// ---------------------------------------------------------------------------
__global__ void k_posbias(const float* __restrict__ table, float* __restrict__ pb) {
  const int c = blockIdx.x * blockDim.x + threadIdx.x;
  const int q = blockIdx.y;
  const int bucket = rel_bucket(c - q);
#pragma unroll
  for (int h = 0; h < kH; ++h)
    pb[((size_t)h * kS + q) * kS + c] = table[bucket * kH + h];
}

// ---------------------------------------------------------------------------
// 5) Scores + bias + softmax -> attn_weights, then fused P@V -> ctx.
//    Block = 8 waves per (b, h, 16-row q-tile); 16x2048 scores in LDS.
//    Q tile staged into LDS via the Tensor Data Mover.
// ---------------------------------------------------------------------------
__global__ void __launch_bounds__(256) k_attn(const __bf16* __restrict__ Qb,
                                              const __bf16* __restrict__ Kb,
                                              const __bf16* __restrict__ Vt,
                                              const float* __restrict__ table,
                                              float* __restrict__ attw,
                                              __bf16* __restrict__ ctx) {
  extern __shared__ float sm[];
  float*  scores = sm;                         // 16 * 2048 f32
  float*  red    = sm + 16 * kS;               // 272 f32
  float*  tbl_s  = red + 272;                  // 512 f32 (bias table)
  __bf16* qtile  = (__bf16*)(tbl_s + 512);     // 16 x 64 bf16

  const int qt = blockIdx.x;
  const int h  = blockIdx.y;
  const int bb = blockIdx.z;
  const int bh = bb * kH + h;
  const int q0 = qt * 16;
  const int tid = threadIdx.x;
  const int w = tid >> 5;
  const int l = tid & 31;

  // --- TDM: DMA the 16x64 bf16 Q tile (row stride 64) into LDS -------------
  if (w == 0) {
    const unsigned long long ga =
        (unsigned long long)(uintptr_t)(Qb + ((size_t)bh * kS + q0) * kHD);
    const unsigned la = (unsigned)(uintptr_t)qtile;
    v4u g0 = { 1u,                                  // count=1 (valid D#)
               la,                                  // lds_addr
               (unsigned)ga,                        // global_addr[31:0]
               (unsigned)(ga >> 32) | (2u << 30) }; // global_addr[56:32] | type=2
    v8i g1 = { (int)(1u << 16),   // data_size = 2 bytes; workgroup_mask = 0
               (int)(64u << 16),  // tensor_dim0 = 64   (bits 79:48)
               (int)(16u << 16),  // tensor_dim1 = 16   (bits 111:80)
               (int)(64u << 16),  // tile_dim0 = 64     (bits 127:112)
               16,                // tile_dim1 = 16
               64,                // tensor_dim0_stride = 64
               0, 0 };
    v4i gz = { 0, 0, 0, 0 };
#if defined(__clang_major__) && (__clang_major__ >= 23)
    v8i gz8 = { 0, 0, 0, 0, 0, 0, 0, 0 };
    __builtin_amdgcn_tensor_load_to_lds(g0, g1, gz, gz, gz8, 0);
#else
    __builtin_amdgcn_tensor_load_to_lds(g0, g1, gz, gz, 0);
#endif
  }
  // Stage the 32x16 bias table into LDS while the TDM runs.
  for (int i = tid; i < 512; i += 256) tbl_s[i] = table[i];
  if (w == 0) __builtin_amdgcn_s_wait_tensorcnt(0);
  __syncthreads();

  const v16bf aq0 = load_a_bf16(qtile, kHD);        // d = 0..31
  const v16bf aq1 = load_a_bf16(qtile + 32, kHD);   // d = 32..63

  for (int j = w; j < kS / 16; j += 8) {
    const __bf16* Ktile = Kb + ((size_t)bh * kS + (size_t)j * 16) * kHD;
    v16bf b0 = load_bt_bf16(Ktile, kHD);
    v16bf b1 = load_bt_bf16(Ktile + 32, kHD);
    v8f acc = {};
    acc = wmma_bf16(aq0, b0, acc);
    acc = wmma_bf16(aq1, b1, acc);
    const int n = l & 15, hv = l >> 4;
#pragma unroll
    for (int r = 0; r < 8; ++r) {
      const int m = r + (hv << 3);
      const int col = j * 16 + n;
      const int bucket = rel_bucket(col - (q0 + m));
      scores[m * kS + col] = acc[r] + tbl_s[bucket * kH + h];
    }
  }
  __syncthreads();

  // --- Row softmax: 16 threads per row ------------------------------------
  const int r = tid >> 4, sub = tid & 15;
  float mx = -3.0e38f;
  for (int c = sub; c < kS; c += 16) mx = fmaxf(mx, scores[r * kS + c]);
  red[r * 16 + sub] = mx;
  __syncthreads();
  if (sub == 0) {
    float m2 = red[r * 16];
#pragma unroll
    for (int t = 1; t < 16; ++t) m2 = fmaxf(m2, red[r * 16 + t]);
    red[256 + r] = m2;
  }
  __syncthreads();
  mx = red[256 + r];
  float sum = 0.f;
  for (int c = sub; c < kS; c += 16) sum += __expf(scores[r * kS + c] - mx);
  red[r * 16 + sub] = sum;
  __syncthreads();
  if (sub == 0) {
    float s2 = 0.f;
#pragma unroll
    for (int t = 0; t < 16; ++t) s2 += red[r * 16 + t];
    red[256 + r] = 1.0f / s2;
  }
  __syncthreads();
  const float rinv = red[256 + r];
  float* outp = attw + ((size_t)bh * kS + q0) * kS;
  for (int c = sub; c < kS; c += 16) {
    const float p = __expf(scores[r * kS + c] - mx) * rinv;
    outp[(size_t)r * kS + c] = p;       // attn_weights output
    scores[r * kS + c] = p;             // keep normalized P in LDS for P@V
  }
  __syncthreads();

  // --- Fused P@V: waves 0..3 each produce one 16x16 d-tile of ctx ---------
  if (w < 4) {
    const __bf16* V = Vt + ((size_t)bh * kHD + w * 16) * kS;
    v8f acc = {};
    for (int k0 = 0; k0 < kS; k0 += 32) {
      v16bf a = load_a_f32(scores + k0, kS);     // LDS f32 -> bf16 fragment
      v16bf b = load_bt_bf16(V + k0, kS);
      acc = wmma_bf16(a, b, acc);
    }
    const int d = w * 16 + (l & 15);
#pragma unroll
    for (int rr = 0; rr < 8; ++rr) {
      const int s = q0 + rr + ((l >> 4) << 3);
      ctx[((size_t)bb * kS + s) * kDM + h * kHD + d] = f2bf(acc[rr]);
    }
  }
}

// ---------------------------------------------------------------------------
// 6) Output projection: ctx[8192x1024] @ Wo -> attn_output f32 (d_out slice)
// ---------------------------------------------------------------------------
__global__ void k_oproj(const __bf16* __restrict__ ctx, const __bf16* __restrict__ Wot,
                        float* __restrict__ out) {
  const int n0 = blockIdx.x * 16;
  const int m0 = blockIdx.y * 16;
  v8f acc = {};
  for (int k0 = 0; k0 < kDM; k0 += 32) {
    v16bf a = load_a_bf16(ctx + (size_t)m0 * kDM + k0, kDM);
    v16bf b = load_bt_bf16(Wot + (size_t)n0 * kDM + k0, kDM);
    acc = wmma_bf16(a, b, acc);
  }
  const int l = threadIdx.x & 31;
  const int n = n0 + (l & 15);
#pragma unroll
  for (int r = 0; r < 8; ++r) {
    const int m = m0 + r + ((l >> 4) << 3);
    out[(size_t)m * kDM + n] = acc[r];
  }
}

// ---------------------------------------------------------------------------
extern "C" void kernel_launch(void* const* d_in, const int* in_sizes, int n_in,
                              void* d_out, int out_size, void* d_ws, size_t ws_size,
                              hipStream_t stream) {
  (void)in_sizes; (void)n_in; (void)out_size; (void)ws_size;
  const float* X   = (const float*)d_in[0];
  const float* Wq  = (const float*)d_in[1];
  const float* Wk  = (const float*)d_in[2];
  const float* Wv  = (const float*)d_in[3];
  const float* Wo  = (const float*)d_in[4];
  const float* tbl = (const float*)d_in[5];

  float* out      = (float*)d_out;
  float* attn_out = out;                                  // [B,S,DM]
  float* posbias  = out + (size_t)kB * kS * kDM;          // [H,S,S]
  float* attw     = posbias + (size_t)kH * kS * kS;       // [B,H,S,S]

  __bf16* p   = (__bf16*)d_ws;
  __bf16* Xb  = p; p += (size_t)kB * kS * kDM;
  __bf16* Wqt = p; p += (size_t)kDM * kDM;
  __bf16* Wkt = p; p += (size_t)kDM * kDM;
  __bf16* Wvt = p; p += (size_t)kDM * kDM;
  __bf16* Wot = p; p += (size_t)kDM * kDM;
  __bf16* Qb  = p; p += (size_t)kB * kH * kS * kHD;
  __bf16* Kb  = p; p += (size_t)kB * kH * kS * kHD;
  __bf16* Vt  = p; p += (size_t)kB * kH * kS * kHD;
  __bf16* ctx = p; p += (size_t)kB * kS * kDM;

  const int nX = kB * kS * kDM;
  k_cvt_bf16<<<nX / 256, 256, 0, stream>>>(X, Xb, nX);

  dim3 tb(32, 8), tg(kDM / 32, kDM / 32);
  k_wt_bf16<<<tg, tb, 0, stream>>>(Wq, Wqt);
  k_wt_bf16<<<tg, tb, 0, stream>>>(Wk, Wkt);
  k_wt_bf16<<<tg, tb, 0, stream>>>(Wv, Wvt);
  k_wt_bf16<<<tg, tb, 0, stream>>>(Wo, Wot);

  k_qkv<<<dim3(kDM / 16, (kB * kS) / 16, 3), 32, 0, stream>>>(Xb, Wqt, Wkt, Wvt,
                                                              Qb, Kb, Vt);
  k_posbias<<<dim3(kS / 256, kS), 256, 0, stream>>>(tbl, posbias);

  // scores (16*2048) + red (272) + table (512) floats + 16x64 bf16 Q tile
  const size_t smem = (size_t)(16 * kS + 272 + 512) * sizeof(float) + 2048;
  k_attn<<<dim3(kS / 16, kH, kB), 256, smem, stream>>>(Qb, Kb, Vt, tbl, attw, ctx);

  k_oproj<<<dim3(kDM / 16, (kB * kS) / 16), 32, 0, stream>>>(ctx, Wot, attn_out);
}
